// ThunderNet_50019189129753
// MI455X (gfx1250) — matrix operand order repaired and checked by reading, not compiled, not run on hardware
//
#include <hip/hip_runtime.h>
#include <hip/hip_bf16.h>
#include <math.h>

// ---------------------------------------------------------------------------
// CDNA5 / gfx1250 WMMA types
// ---------------------------------------------------------------------------
typedef __attribute__((ext_vector_type(16))) _Float16     v16h;
typedef __attribute__((ext_vector_type(8)))  float        v8f;
typedef __attribute__((ext_vector_type(4)))  unsigned int v4u;

// exact pointee type expected by the async-LDS builtins (per hipcc diagnostic)
typedef int v4i_gcc __attribute__((vector_size(16)));
#define GLOBAL_AS __attribute__((address_space(1)))
#define LDS_AS    __attribute__((address_space(3)))

union FragH { v16h h; v4u u[2]; };

#ifndef __has_builtin
#define __has_builtin(x) 0
#endif
#if __has_builtin(__builtin_amdgcn_global_load_async_to_lds_b128) && \
    __has_builtin(__builtin_amdgcn_s_wait_asynccnt)
#define HAS_ASYNC_LDS 1
#else
#define HAS_ASYNC_LDS 0
#endif

// ---------------------------------------------------------------------------
// WMMA GEMM:  C[M x N] = A[Mp x Kp] * B[Kp x Np]  (+bias[m], opt ReLU)
//  A  : fp16 row-major (Mp x Kp), zero padded, shared across batch
//  Bt : fp16 "N-major": row n holds its K values contiguously (Np x Kp)
//  C  : fp32 row-major, leading dim ldC; only m<Mstore, n<Nstore stored
//
//  Workgroup = 128 threads (4 waves). Each block owns one 16-row M tile and a
//  64-wide N panel; the A tile is staged into LDS (async global->LDS on CDNA5,
//  ASYNCcnt + s_wait_asynccnt) and shared by all 4 waves; each wave computes a
//  16x16 tile per panel column via v_wmma_f32_16x16x32_f16.
//  Fragment layouts per CDNA5 ISA.
// ---------------------------------------------------------------------------
#define KC 128   // K chunk staged in LDS (halves); Kp is a multiple of 32

__global__ __launch_bounds__(128)
void wmma_gemm(const _Float16* __restrict__ A,
               const _Float16* __restrict__ Bt,
               const float*    __restrict__ bias,
               float*          __restrict__ C,
               int Kp, int ldC, int Mstore, int Nstore,
               long strideB, long strideC, int act)
{
    __shared__ __align__(16) _Float16 sA[16 * KC];

    const int tid  = threadIdx.x;
    const int lane = tid & 31;
    const int wave = tid >> 5;
    const int hi   = lane >> 4;   // 0: low K half, 1: high K half
    const int lo   = lane & 15;

    const int mbase = blockIdx.y * 16;
    const int nbase = blockIdx.x * 64 + wave * 16;

    const _Float16* Ablk = A + (long)mbase * Kp;
    const _Float16* brow = Bt + (long)blockIdx.z * strideB
                              + (long)(nbase + lo) * Kp;

    v8f acc = {};

    for (int k0 = 0; k0 < Kp; k0 += KC) {
        const int cur = (Kp - k0) < KC ? (Kp - k0) : KC;  // multiple of 32
        const int vecPerRow = cur >> 3;                   // b128 vectors / row
        const int nVec = vecPerRow << 4;                  // 16 rows

        // --- stage A[mbase..mbase+15][k0..k0+cur) into LDS ---
        for (int i = tid; i < nVec; i += 128) {
            int row = i / vecPerRow;
            int vo  = i - row * vecPerRow;
            const _Float16* src = Ablk + (long)row * Kp + k0 + (vo << 3);
            _Float16*       dst = sA + row * cur + (vo << 3);
#if HAS_ASYNC_LDS
            __builtin_amdgcn_global_load_async_to_lds_b128(
                (GLOBAL_AS v4i_gcc*)src,
                (LDS_AS    v4i_gcc*)dst, 0, 0);
#else
            *(v4u*)dst = *(const v4u*)src;
#endif
        }
#if HAS_ASYNC_LDS
        __builtin_amdgcn_s_wait_asynccnt(0);
#endif
        __syncthreads();

        // --- compute over this K chunk ---
        const int ksteps = cur >> 5;
        for (int ks = 0; ks < ksteps; ++ks) {
            FragH a, b;
            const _Float16* pa = sA + lo * cur + (ks << 5) + (hi << 3);
            a.u[0] = *(const v4u*)(pa);        // A: K {0..7}  (+8 if hi)  [LDS]
            a.u[1] = *(const v4u*)(pa + 16);   // A: K {16..23}(+8 if hi)  [LDS]
            const _Float16* pb = brow + k0 + (ks << 5) + (hi << 3);
            b.u[0] = *(const v4u*)(pb);        // B: K {0..7}  (+8 if hi)
            b.u[1] = *(const v4u*)(pb + 16);   // B: K {16..23}(+8 if hi)
            __builtin_prefetch(pb + KC, 0, 1); // -> global_prefetch_b8
            acc = __builtin_amdgcn_wmma_f32_16x16x32_f16(
                /*neg_a=*/false, a.h, /*neg_b=*/false, b.h,
                /*c_mod=*/(short)0, acc, /*reuse_a=*/false, /*reuse_b=*/false);
        }
        __syncthreads();
    }

    float* Ci = C + (long)blockIdx.z * strideC;
    const int n = nbase + lo;
    if (n < Nstore) {
        #pragma unroll
        for (int v = 0; v < 8; ++v) {
            int m = mbase + (hi << 3) + v;     // C/D layout per ISA
            if (m < Mstore) {
                float val = acc[v];
                if (bias) val += bias[m];
                if (act)  val = fmaxf(val, 0.f);
                Ci[(long)m * ldC + n] = val;
            }
        }
    }
}

// ---------------------------------------------------------------------------
// Packing helpers (fp32 -> zero-padded fp16)
// ---------------------------------------------------------------------------
__global__ void pack_f16(const float* __restrict__ src, _Float16* __restrict__ dst,
                         int Rs, int Cs, int Mp, int Kp, long sSrc, long sDst)
{
    long idx = (long)blockIdx.x * blockDim.x + threadIdx.x;
    long total = (long)Mp * Kp;
    if (idx >= total) return;
    int k = (int)(idx % Kp), m = (int)(idx / Kp);
    float v = 0.f;
    if (m < Rs && k < Cs)
        v = src[(long)blockIdx.z * sSrc + (long)m * Cs + k];
    dst[(long)blockIdx.z * sDst + idx] = (_Float16)v;
}

// transpose pack: dst[m*Kp+k] = src[k*Cs+m]   (src is Rs x Cs)
__global__ void packT_f16(const float* __restrict__ src, _Float16* __restrict__ dst,
                          int Rs, int Cs, int Mp, int Kp, long sSrc, long sDst)
{
    long idx = (long)blockIdx.x * blockDim.x + threadIdx.x;
    long total = (long)Mp * Kp;
    if (idx >= total) return;
    int k = (int)(idx % Kp), m = (int)(idx / Kp);
    float v = 0.f;
    if (m < Cs && k < Rs)
        v = src[(long)blockIdx.z * sSrc + (long)k * Cs + m];
    dst[(long)blockIdx.z * sDst + idx] = (_Float16)v;
}

// im2col for 3x3 stride-2 SAME (pad lo=0, hi=1) -> Bt layout (Np x Kp) fp16
__global__ void im2col3x3s2(const float* __restrict__ src, _Float16* __restrict__ dst,
                            int Cin, int Hin, int Win, int Hout, int Wout,
                            int Kp, int Np, long sSrc, long sDst)
{
    long idx = (long)blockIdx.x * blockDim.x + threadIdx.x;
    long total = (long)Np * Kp;
    if (idx >= total) return;
    int k = (int)(idx % Kp), n = (int)(idx / Kp);
    float v = 0.f;
    if (n < Hout * Wout && k < Cin * 9) {
        int ci = k / 9, t = k % 9, ky = t / 3, kx = t % 3;
        int oy = n / Wout, ox = n % Wout;
        int iy = oy * 2 + ky, ix = ox * 2 + kx;
        if (iy < Hin && ix < Win)
            v = src[(long)blockIdx.z * sSrc + ((long)ci * Hin + iy) * Win + ix];
    }
    dst[(long)blockIdx.z * sDst + idx] = (_Float16)v;
}

// ---------------------------------------------------------------------------
// Scalar backbone pieces
// ---------------------------------------------------------------------------
__global__ void conv1_relu(const float* __restrict__ x, const float* __restrict__ w,
                           const float* __restrict__ b, float* __restrict__ y)
{
    long idx = (long)blockIdx.x * blockDim.x + threadIdx.x;
    const long total = 8L * 24 * 160 * 160;
    if (idx >= total) return;
    int p = (int)(idx % 25600); int t = (int)(idx / 25600);
    int co = t % 24, img = t / 24;
    int oy = p / 160, ox = p % 160;
    float acc = b[co];
    for (int ci = 0; ci < 3; ++ci)
        for (int ky = 0; ky < 3; ++ky) {
            int iy = oy * 2 + ky;
            if (iy >= 320) continue;
            for (int kx = 0; kx < 3; ++kx) {
                int ix = ox * 2 + kx;
                if (ix >= 320) continue;
                acc += x[((long)img * 3 + ci) * 102400 + iy * 320 + ix]
                     * w[(co * 3 + ci) * 9 + ky * 3 + kx];
            }
        }
    y[idx] = fmaxf(acc, 0.f);
}

__global__ void maxpool3s2(const float* __restrict__ in, float* __restrict__ outp)
{
    long idx = (long)blockIdx.x * blockDim.x + threadIdx.x;
    const long total = 8L * 24 * 80 * 80;
    if (idx >= total) return;
    int p = (int)(idx % 6400); int t = (int)(idx / 6400);
    int c = t % 24, img = t / 24;
    int oy = p / 80, ox = p % 80;
    float m = -INFINITY;
    for (int ky = 0; ky < 3; ++ky) {
        int iy = oy * 2 + ky;
        if (iy >= 160) continue;
        for (int kx = 0; kx < 3; ++kx) {
            int ix = ox * 2 + kx;
            if (ix >= 160) continue;
            m = fmaxf(m, in[((long)img * 24 + c) * 25600 + iy * 160 + ix]);
        }
    }
    outp[idx] = m;
}

__global__ void cglb_k(const float* __restrict__ c5, float* __restrict__ cglb)
{
    int idx = blockIdx.x * blockDim.x + threadIdx.x;
    if (idx >= 8 * 528) return;
    const float* p = c5 + (long)idx * 100;
    float s = 0.f;
    for (int i = 0; i < 100; ++i) s += p[i];
    cglb[idx] = s * 0.01f;
}

__global__ void glbvec_k(const float* __restrict__ cglb, const float* __restrict__ wg,
                         const float* __restrict__ bg, float* __restrict__ gv)
{
    int idx = blockIdx.x * blockDim.x + threadIdx.x;
    if (idx >= 8 * 245) return;
    int c = idx % 245, img = idx / 245;
    float s = bg[c];
    for (int k = 0; k < 528; ++k) s += cglb[img * 528 + k] * wg[k * 245 + c];
    gv[idx] = s;
}

__global__ void fcem_comb(const float* __restrict__ a4, const float* __restrict__ a5,
                          const float* __restrict__ gv, float* __restrict__ fcem)
{
    long idx = (long)blockIdx.x * blockDim.x + threadIdx.x;
    const long total = 8L * 245 * 400;
    if (idx >= total) return;
    int pos = (int)(idx % 400); int t = (int)(idx / 400);
    int c = t % 245, img = t / 245;
    int y = pos / 20, x = pos % 20;
    fcem[idx] = a4[idx]
              + a5[((long)img * 245 + c) * 100 + (y >> 1) * 10 + (x >> 1)]
              + gv[img * 245 + c];
}

// depthwise 5x5 SAME (pad 2) + relu
__global__ void dw5_relu(const float* __restrict__ f, const float* __restrict__ dw,
                         const float* __restrict__ db, float* __restrict__ t)
{
    long idx = (long)blockIdx.x * blockDim.x + threadIdx.x;
    const long total = 8L * 245 * 400;
    if (idx >= total) return;
    int pos = (int)(idx % 400); int q = (int)(idx / 400);
    int c = q % 245, img = q / 245;
    int y = pos / 20, x = pos % 20;
    float acc = db[c];
    for (int ky = 0; ky < 5; ++ky) {
        int iy = y + ky - 2;
        if (iy < 0 || iy >= 20) continue;
        for (int kx = 0; kx < 5; ++kx) {
            int ix = x + kx - 2;
            if (ix < 0 || ix >= 20) continue;
            acc += f[((long)img * 245 + c) * 400 + iy * 20 + ix] * dw[c * 25 + ky * 5 + kx];
        }
    }
    t[idx] = fmaxf(acc, 0.f);
}

__global__ void sigmul(const float* __restrict__ f, const float* __restrict__ s,
                       float* __restrict__ o)
{
    long idx = (long)blockIdx.x * blockDim.x + threadIdx.x;
    const long total = 8L * 245 * 400;
    if (idx >= total) return;
    o[idx] = f[idx] * (1.f / (1.f + expf(-s[idx])));
}

// ---------------------------------------------------------------------------
// RPN post-processing
// ---------------------------------------------------------------------------
__global__ void rpn_decode(const float* __restrict__ cls, const float* __restrict__ reg,
                           float* __restrict__ boxes, float* __restrict__ ss,
                           int* __restrict__ si)
{
    long idx = (long)blockIdx.x * blockDim.x + threadIdx.x;
    const long total = 8L * 16384;
    if (idx >= total) return;
    int r = (int)(idx % 16384); int img = (int)(idx / 16384);
    if (r >= 10000) { ss[idx] = -3.0e38f; si[idx] = r; return; }
    int pos = r / 25, a = r % 25;
    const float* ci = cls + (long)img * 50 * 400;
    float l0 = ci[(a * 2) * 400 + pos], l1 = ci[(a * 2 + 1) * 400 + pos];
    ss[idx] = 1.f / (1.f + expf(l0 - l1));
    si[idx] = r;
    const float* ri = reg + (long)img * 100 * 400;
    float d0 = ri[(a * 4 + 0) * 400 + pos];
    float d1 = ri[(a * 4 + 1) * 400 + pos];
    float d2 = ri[(a * 4 + 2) * 400 + pos];
    float d3 = ri[(a * 4 + 3) * 400 + pos];
    const float ratios[5] = {0.5f, 0.75f, 1.f, 4.f / 3.f, 2.f};
    int sidx2 = a / 5, ridx2 = a % 5;
    float scale = 32.f * (float)(1 << sidx2);
    float sq = sqrtf(ratios[ridx2]);
    float aw = scale / sq, ah = scale * sq;
    int y = pos / 20, x = pos % 20;
    float acx = (x + 0.5f) * 16.f, acy = (y + 0.5f) * 16.f;
    float cx = d0 * aw + acx, cy = d1 * ah + acy;
    float w = expf(d2) * aw, h = expf(d3) * ah;
    float* bp = boxes + ((long)img * 10000 + r) * 4;
    bp[0] = fminf(fmaxf(cx - 0.5f * w, 0.f), 319.f);
    bp[1] = fminf(fmaxf(cy - 0.5f * h, 0.f), 319.f);
    bp[2] = fminf(fmaxf(cx + 0.5f * w, 0.f), 319.f);
    bp[3] = fminf(fmaxf(cy + 0.5f * h, 0.f), 319.f);
}

// single-block bitonic sort (descending, tie -> smaller index first), n=16384
__global__ __launch_bounds__(1024) void bitonic16k(float* __restrict__ ss,
                                                   int* __restrict__ si)
{
    const int n = 16384;
    float* s  = ss + (long)blockIdx.x * n;
    int*   id = si + (long)blockIdx.x * n;
    for (int k = 2; k <= n; k <<= 1) {
        for (int j = k >> 1; j > 0; j >>= 1) {
            for (int t = threadIdx.x; t < n; t += 1024) {
                int ixj = t ^ j;
                if (ixj > t) {
                    bool up = ((t & k) == 0);
                    float a = s[t], b = s[ixj];
                    int ia = id[t], ib = id[ixj];
                    bool aWins = (a > b) || (a == b && ia < ib);
                    if (up ? !aWins : aWins) {
                        s[t] = b; s[ixj] = a; id[t] = ib; id[ixj] = ia;
                    }
                }
            }
            __syncthreads();
        }
    }
}

// single-block NMS over top-1000, then emit 200 rois (kept-first stable order)
__global__ __launch_bounds__(256) void nms_rois(const float* __restrict__ boxes,
                                                const int* __restrict__ si,
                                                float* __restrict__ rois)
{
    __shared__ float4 bx[1000];
    __shared__ float  ar[1000];
    __shared__ int    supp[1000];
    __shared__ int    curKeep;
    int img = blockIdx.x;
    for (int r = threadIdx.x; r < 1000; r += 256) {
        int id = si[(long)img * 16384 + r];
        const float* bp = boxes + ((long)img * 10000 + id) * 4;
        float4 bb = make_float4(bp[0], bp[1], bp[2], bp[3]);
        bx[r] = bb;
        ar[r] = (bb.z - bb.x) * (bb.w - bb.y);
        supp[r] = 0;
    }
    __syncthreads();
    for (int i = 0; i < 1000; ++i) {
        if (threadIdx.x == 0) curKeep = supp[i] ? 0 : 1;
        __syncthreads();
        if (curKeep) {
            float4 bi = bx[i];
            float ai = ar[i];
            for (int j = i + 1 + threadIdx.x; j < 1000; j += 256) {
                float4 bj = bx[j];
                float ix1 = fmaxf(bi.x, bj.x), iy1 = fmaxf(bi.y, bj.y);
                float ix2 = fminf(bi.z, bj.z), iy2 = fminf(bi.w, bj.w);
                float inter = fmaxf(ix2 - ix1, 0.f) * fmaxf(iy2 - iy1, 0.f);
                float iou = inter / (ai + ar[j] - inter + 1e-9f);
                if (iou > 0.7f) supp[j] = 1;
            }
        }
        __syncthreads();
    }
    if (threadIdx.x == 0) {
        float* rp = rois + (long)img * 200 * 4;
        int cnt = 0;
        for (int r = 0; r < 1000 && cnt < 200; ++r)
            if (!supp[r]) {
                float4 bb = bx[r];
                rp[cnt * 4 + 0] = bb.x; rp[cnt * 4 + 1] = bb.y;
                rp[cnt * 4 + 2] = bb.z; rp[cnt * 4 + 3] = bb.w;
                ++cnt;
            }
        for (int r = 0; r < 1000 && cnt < 200; ++r)
            if (supp[r]) {
                float4 bb = bx[r];
                rp[cnt * 4 + 0] = bb.x; rp[cnt * 4 + 1] = bb.y;
                rp[cnt * 4 + 2] = bb.z; rp[cnt * 4 + 3] = bb.w;
                ++cnt;
            }
    }
}

// position-sensitive ROI align -> fp16 feature rows (FC GEMM B operand)
__global__ void psroi_feat(const float* __restrict__ fsam, const float* __restrict__ rois,
                           _Float16* __restrict__ feat)
{
    int r = blockIdx.x;       // roi 0..199
    int img = blockIdx.y;     // image 0..7
    int f = threadIdx.x;      // feature (alpha,gi,gj) == channel index
    const float* rp = rois + ((long)img * 200 + r) * 4;
    float x1 = rp[0] / 16.f, y1 = rp[1] / 16.f;
    float x2 = rp[2] / 16.f, y2 = rp[3] / 16.f;
    float bw = fmaxf(x2 - x1, 0.1f) / 7.f;
    float bh = fmaxf(y2 - y1, 0.1f) / 7.f;
    if (f < 245) {
        int rem = f % 49, gi = rem / 7, gj = rem % 7;
        float ysr = y1 + (gi + 0.5f) * bh;
        float xsr = x1 + (gj + 0.5f) * bw;
        float y = fminf(fmaxf(ysr, 0.f), 19.f);
        float x = fminf(fmaxf(xsr, 0.f), 19.f);
        int y0 = (int)floorf(y); y0 = y0 < 0 ? 0 : (y0 > 18 ? 18 : y0);
        int x0 = (int)floorf(x); x0 = x0 < 0 ? 0 : (x0 > 18 ? 18 : x0);
        float wy = y - (float)y0, wx = x - (float)x0;
        const float* fm = fsam + ((long)img * 245 + f) * 400;
        float v00 = fm[y0 * 20 + x0],       v01 = fm[y0 * 20 + x0 + 1];
        float v10 = fm[(y0 + 1) * 20 + x0], v11 = fm[(y0 + 1) * 20 + x0 + 1];
        float val = v00 * (1 - wy) * (1 - wx) + v01 * (1 - wy) * wx
                  + v10 * wy * (1 - wx)       + v11 * wy * wx;
        feat[((long)img * 256 + r) * 256 + f] = (_Float16)val;
    }
}

__global__ void gather_out(const float* __restrict__ clsT, const float* __restrict__ regT,
                           const float* __restrict__ rois, float* __restrict__ out)
{
    long idx = (long)blockIdx.x * blockDim.x + threadIdx.x;
    if (idx >= 46400) return;
    if (idx < 33600) {
        int img = (int)(idx / (200 * 21));
        int rem = (int)(idx % (200 * 21));
        int r = rem / 21, c = rem % 21;
        out[idx] = clsT[(long)img * 21 * 256 + c * 256 + r];
    } else if (idx < 40000) {
        long j = idx - 33600;
        int img = (int)(j / 800);
        int rem = (int)(j % 800);
        int r = rem / 4, c = rem % 4;
        out[idx] = regT[(long)img * 4 * 256 + c * 256 + r];
    } else {
        out[idx] = rois[idx - 40000];
    }
}

// ---------------------------------------------------------------------------
// Host orchestration
// ---------------------------------------------------------------------------
extern "C" void kernel_launch(void* const* d_in, const int* in_sizes, int n_in,
                              void* d_out, int out_size, void* d_ws, size_t ws_size,
                              hipStream_t stream)
{
    (void)in_sizes; (void)n_in; (void)out_size; (void)ws_size;
    const float* x      = (const float*)d_in[0];
    const float* w1     = (const float*)d_in[1];
    const float* b1     = (const float*)d_in[2];
    const float* w2     = (const float*)d_in[3];
    const float* b2     = (const float*)d_in[4];
    const float* w3     = (const float*)d_in[5];
    const float* b3     = (const float*)d_in[6];
    const float* w4     = (const float*)d_in[7];
    const float* b4     = (const float*)d_in[8];
    const float* cem_w4 = (const float*)d_in[9];
    const float* cem_b4 = (const float*)d_in[10];
    const float* cem_w5 = (const float*)d_in[11];
    const float* cem_b5 = (const float*)d_in[12];
    const float* cem_wg = (const float*)d_in[13];
    const float* cem_bg = (const float*)d_in[14];
    const float* rpn_dw = (const float*)d_in[15];
    const float* rpn_dwb= (const float*)d_in[16];
    const float* rpn_pw = (const float*)d_in[17];
    const float* rpn_pwb= (const float*)d_in[18];
    const float* cls_w  = (const float*)d_in[19];
    const float* cls_b  = (const float*)d_in[20];
    const float* reg_w  = (const float*)d_in[21];
    const float* reg_b  = (const float*)d_in[22];
    const float* sam_w  = (const float*)d_in[23];
    const float* sam_b  = (const float*)d_in[24];
    const float* fc_w   = (const float*)d_in[25];
    const float* fc_b   = (const float*)d_in[26];
    const float* cls_fw = (const float*)d_in[27];
    const float* cls_fb = (const float*)d_in[28];
    const float* reg_fw = (const float*)d_in[29];
    const float* reg_fb = (const float*)d_in[30];
    float* out = (float*)d_out;

    char* base = (char*)d_ws;
    size_t off = 0;
    auto alloc = [&](size_t bytes) -> void* {
        off = (off + 255) & ~(size_t)255;
        void* p = base + off;
        off += bytes;
        return p;
    };
    auto AF = [&](long n) { return (float*)alloc((size_t)n * 4); };
    auto AH = [&](long n) { return (_Float16*)alloc((size_t)n * 2); };

    float*    c1    = AF(8L * 24 * 160 * 160);
    float*    p1    = AF(8L * 24 * 80 * 80);
    _Float16* w2h   = AH(144L * 224);
    _Float16* bt2   = AH(8L * 1600 * 224);
    float*    c3    = AF(8L * 132 * 1600);
    _Float16* w3h   = AH(272L * 1216);
    _Float16* bt3   = AH(8L * 448 * 1216);
    float*    c4    = AF(8L * 264 * 400);
    _Float16* w4h   = AH(528L * 2400);
    _Float16* bt4   = AH(8L * 128 * 2400);
    float*    c5    = AF(8L * 528 * 100);
    float*    cglb  = AF(8L * 528);
    float*    glbv  = AF(8L * 245);
    _Float16* cw4h  = AH(256L * 288);
    _Float16* cw5h  = AH(256L * 544);
    _Float16* c4t   = AH(8L * 448 * 288);
    _Float16* c5t   = AH(8L * 128 * 544);
    float*    cem4o = AF(8L * 245 * 400);
    float*    cem5o = AF(8L * 245 * 100);
    float*    fcem  = AF(8L * 245 * 400);
    float*    tbuf  = AF(8L * 245 * 400);
    _Float16* tt    = AH(8L * 448 * 256);
    _Float16* pwh   = AH(256L * 256);
    float*    frpn  = AF(8L * 256 * 400);
    _Float16* frt   = AH(8L * 448 * 256);
    _Float16* clswh = AH(64L * 256);
    _Float16* regwh = AH(112L * 256);
    _Float16* samwh = AH(256L * 256);
    float*    rpncls= AF(8L * 50 * 400);
    float*    rpnreg= AF(8L * 100 * 400);
    float*    samo  = AF(8L * 245 * 400);
    float*    fsam  = AF(8L * 245 * 400);
    float*    boxes = AF(8L * 10000 * 4);
    float*    ss    = AF(8L * 16384);
    int*      sidx  = (int*)alloc(8L * 16384 * 4);
    float*    roisb = AF(8L * 200 * 4);
    _Float16* feath = AH(8L * 256 * 256);
    _Float16* fcwh  = AH(1024L * 256);
    float*    hT    = AF(8L * 1024 * 256);
    _Float16* hh    = AH(8L * 256 * 1024);
    _Float16* cfwh  = AH(32L * 1024);
    _Float16* rfwh  = AH(16L * 1024);
    float*    clsT  = AF(8L * 21 * 256);
    float*    regT  = AF(8L * 4 * 256);

    // zero workspace (covers all fp16 pads); graph-capturable stream op
    (void)hipMemsetAsync(d_ws, 0, off, stream);

    const int TB = 256;
    auto G1 = [](long n) { return dim3((unsigned)((n + 255) / 256)); };

    auto pack = [&](const float* s, _Float16* d, int Rs, int Cs, int Mp, int Kp) {
        long tot = (long)Mp * Kp;
        pack_f16<<<dim3((unsigned)((tot + TB - 1) / TB), 1, 1), TB, 0, stream>>>(
            s, d, Rs, Cs, Mp, Kp, 0, 0);
    };
    auto packT = [&](const float* s, _Float16* d, int Rs, int Cs, int Mp, int Kp,
                     long sS, long sD, int batch) {
        long tot = (long)Mp * Kp;
        packT_f16<<<dim3((unsigned)((tot + TB - 1) / TB), 1, batch), TB, 0, stream>>>(
            s, d, Rs, Cs, Mp, Kp, sS, sD);
    };
    auto gemm = [&](const _Float16* A, const _Float16* Bt, const float* bias, float* C,
                    int Mp, int Np, int Kp, int Mst, int Nst, int ldC,
                    long sB, long sC, int act) {
        wmma_gemm<<<dim3(Np / 64, Mp / 16, 8), dim3(128), 0, stream>>>(
            A, Bt, bias, C, Kp, ldC, Mst, Nst, sB, sC, act);
    };
    auto im2col = [&](const float* s, _Float16* d, int Cin, int Hin, int Win,
                      int Hout, int Wout, int Kp, int Np, long sS, long sD) {
        long tot = (long)Np * Kp;
        im2col3x3s2<<<dim3((unsigned)((tot + TB - 1) / TB), 1, 8), TB, 0, stream>>>(
            s, d, Cin, Hin, Win, Hout, Wout, Kp, Np, sS, sD);
    };

    // backbone stem
    conv1_relu<<<G1(8L * 24 * 160 * 160), TB, 0, stream>>>(x, w1, b1, c1);
    maxpool3s2<<<G1(8L * 24 * 80 * 80), TB, 0, stream>>>(c1, p1);

    // weight packing (fp32 -> padded fp16)
    pack(w2, w2h, 132, 216, 144, 224);
    pack(w3, w3h, 264, 1188, 272, 1216);
    pack(w4, w4h, 528, 2376, 528, 2400);
    pack(cem_w4, cw4h, 245, 264, 256, 288);
    pack(cem_w5, cw5h, 245, 528, 256, 544);
    pack(rpn_pw, pwh, 256, 245, 256, 256);
    pack(cls_w, clswh, 50, 256, 64, 256);
    pack(reg_w, regwh, 100, 256, 112, 256);
    pack(sam_w, samwh, 245, 256, 256, 256);
    packT(fc_w,   fcwh, 245, 1024, 1024, 256, 0, 0, 1);
    packT(cls_fw, cfwh, 1024, 21,  32, 1024, 0, 0, 1);
    packT(reg_fw, rfwh, 1024, 4,   16, 1024, 0, 0, 1);

    // conv2/3/4 via implicit GEMM (WMMA, async-LDS staged A)
    im2col(p1, bt2, 24, 80, 80, 40, 40, 224, 1600, 24L * 80 * 80, 1600L * 224);
    gemm(w2h, bt2, b2, c3, 144, 1600, 224, 132, 1600, 1600, 1600L * 224, 132L * 1600, 1);
    im2col(c3, bt3, 132, 40, 40, 20, 20, 1216, 448, 132L * 1600, 448L * 1216);
    gemm(w3h, bt3, b3, c4, 272, 448, 1216, 264, 400, 400, 448L * 1216, 264L * 400, 1);
    im2col(c4, bt4, 264, 20, 20, 10, 10, 2400, 128, 264L * 400, 128L * 2400);
    gemm(w4h, bt4, b4, c5, 528, 128, 2400, 528, 100, 100, 128L * 2400, 528L * 100, 1);

    // CEM
    cglb_k<<<G1(8L * 528), TB, 0, stream>>>(c5, cglb);
    glbvec_k<<<G1(8L * 245), TB, 0, stream>>>(cglb, cem_wg, cem_bg, glbv);
    packT(c4, c4t, 264, 400, 448, 288, 264L * 400, 448L * 288, 8);
    gemm(cw4h, c4t, cem_b4, cem4o, 256, 448, 288, 245, 400, 400, 448L * 288, 245L * 400, 0);
    packT(c5, c5t, 528, 100, 128, 544, 528L * 100, 128L * 544, 8);
    gemm(cw5h, c5t, cem_b5, cem5o, 256, 128, 544, 245, 100, 100, 128L * 544, 245L * 100, 0);
    fcem_comb<<<G1(8L * 245 * 400), TB, 0, stream>>>(cem4o, cem5o, glbv, fcem);

    // RPN
    dw5_relu<<<G1(8L * 245 * 400), TB, 0, stream>>>(fcem, rpn_dw, rpn_dwb, tbuf);
    packT(tbuf, tt, 245, 400, 448, 256, 245L * 400, 448L * 256, 8);
    gemm(pwh, tt, rpn_pwb, frpn, 256, 448, 256, 256, 400, 400, 448L * 256, 256L * 400, 1);
    packT(frpn, frt, 256, 400, 448, 256, 256L * 400, 448L * 256, 8);
    gemm(clswh, frt, cls_b, rpncls, 64, 448, 256, 50, 400, 400, 448L * 256, 50L * 400, 0);
    gemm(regwh, frt, reg_b, rpnreg, 112, 448, 256, 100, 400, 400, 448L * 256, 100L * 400, 0);
    gemm(samwh, frt, sam_b, samo, 256, 448, 256, 245, 400, 400, 448L * 256, 245L * 400, 0);
    sigmul<<<G1(8L * 245 * 400), TB, 0, stream>>>(fcem, samo, fsam);

    // proposals: decode -> sort -> NMS -> rois
    rpn_decode<<<G1(8L * 16384), TB, 0, stream>>>(rpncls, rpnreg, boxes, ss, sidx);
    bitonic16k<<<dim3(8), dim3(1024), 0, stream>>>(ss, sidx);
    nms_rois<<<dim3(8), dim3(256), 0, stream>>>(boxes, sidx, roisb);

    // heads
    psroi_feat<<<dim3(200, 8), dim3(256), 0, stream>>>(fsam, roisb, feath);
    gemm(fcwh, feath, fc_b, hT, 1024, 256, 256, 1024, 256, 256, 256L * 256, 1024L * 256, 1);
    packT(hT, hh, 1024, 256, 256, 1024, 1024L * 256, 256L * 1024, 8);
    gemm(cfwh, hh, cls_fb, clsT, 32, 256, 1024, 21, 256, 256, 256L * 1024, 21L * 256, 0);
    gemm(rfwh, hh, reg_fb, regT, 16, 256, 1024, 4, 256, 256, 256L * 1024, 4L * 256, 0);

    gather_out<<<G1(46400), TB, 0, stream>>>(clsT, regT, roisb, out);
}
